// NonAutoRegressive_17334488006945
// MI455X (gfx1250) — matrix operand order repaired and checked by reading, not compiled, not run on hardware
//
#include <hip/hip_runtime.h>
#include <cmath>

// ---------------- problem constants ----------------
#define Bz      4
#define TTEXT   128
#define TAUDIO  512
#define Dm      1024
#define NHEADS  16
#define DH      64
#define DFFm    4096
#define NLAY    4
#define NCH     8
#define AVOC    1024
#define ENRn    151              // 2*75+1
#define LMAXn   792              // 128+151+512+1
#define MTOK    (LMAXn*Bz)       // 3168
#define SPAD    800              // 792 padded to 50*16
#define LDST    40               // LDS row stride in ushorts (16B-aligned chunks)

typedef __attribute__((ext_vector_type(16))) __bf16 v16bf;
typedef __attribute__((ext_vector_type(8)))  float  v8f;

union Frag { v16bf v; unsigned short u[16]; };

__device__ __forceinline__ unsigned short f2bf(float f) {
  union { float f; unsigned u; } x; x.f = f;
  unsigned r = x.u + 0x7FFFu + ((x.u >> 16) & 1u);   // round-to-nearest-even
  return (unsigned short)(r >> 16);
}

// 16-byte async copy global -> LDS (ASYNCcnt-tracked, CDNA5 path)
__device__ __forceinline__ void async_copy16(unsigned lds_off, const unsigned short* gptr) {
  asm volatile("global_load_async_to_lds_b128 %0, %1, off"
               :: "v"(lds_off), "v"(gptr) : "memory");
}
__device__ __forceinline__ void async_wait_le4() {   // previous tile done, next tile in flight
  asm volatile("s_wait_asynccnt 4" ::: "memory");
}
__device__ __forceinline__ void async_wait_le0() {
  asm volatile("s_wait_asynccnt 0" ::: "memory");
}

// ---------------- f32 -> bf16 bulk converter (weights, once per call) ----------------
__global__ __launch_bounds__(256)
void nar_cvt_kernel(const float* __restrict__ in, unsigned short* __restrict__ out) {
  long i = ((long)blockIdx.x*256 + threadIdx.x)*8;
  const float4* p = (const float4*)(in + i);
  float4 a = p[0], b = p[1];
  uint4 o;
  o.x = (unsigned)f2bf(a.x) | ((unsigned)f2bf(a.y) << 16);
  o.y = (unsigned)f2bf(a.z) | ((unsigned)f2bf(a.w) << 16);
  o.z = (unsigned)f2bf(b.x) | ((unsigned)f2bf(b.y) << 16);
  o.w = (unsigned)f2bf(b.z) | ((unsigned)f2bf(b.w) << 16);
  *(uint4*)(out + i) = o;
}

// ---------------- embedding + PE -> x (f32), xbf, membf (bf16) ----------------
__global__ __launch_bounds__(256)
void nar_embed_kernel(const int* __restrict__ text, const int* __restrict__ audio,
                      const int* __restrict__ enr, const int* __restrict__ text_len,
                      const int* __restrict__ audio_len, const int* __restrict__ idxp,
                      const float* __restrict__ text_table,
                      const float* __restrict__ audio_tabs,
                      const float* __restrict__ index_tab,
                      float* __restrict__ x, unsigned short* __restrict__ xbf,
                      unsigned short* __restrict__ membf)
{
  int tok = blockIdx.x;            // tok = l*B + b
  int l = tok / Bz, b = tok % Bz;
  int tl = text_len[b];
  int b1 = tl + ENRn;
  int b2 = b1 + audio_len[b];
  int idx = *idxp;
  const float logdiv = -logf(10000.f) / (float)Dm;
  for (int d = threadIdx.x; d < Dm; d += blockDim.x) {
    float v = 0.f;
    int pos = -1;
    if (l < tl) {
      int t = text[b*TTEXT + l];
      v = text_table[(long)t*Dm + d];
      pos = l;
    } else if (l < b1) {
      int pp = l - tl;
      for (int i = 0; i < NCH; ++i) {
        int t = enr[((long)b*NCH + i)*ENRn + pp];
        v += audio_tabs[((long)i*AVOC + t)*Dm + d];
      }
      pos = pp;
    } else if (l < b2) {
      int pp = l - b1;
      for (int i = 0; i < idx; ++i) {
        int t = audio[((long)b*NCH + i)*TAUDIO + pp];
        v += audio_tabs[((long)i*AVOC + t)*Dm + d];
      }
      pos = pp;
    } else if (l == b2) {
      v = index_tab[idx*Dm + d];
      pos = 0;
    }
    if (pos >= 0) {
      float div = expf((float)(d & ~1) * logdiv);
      float arg = (float)pos * div;
      v += (d & 1) ? cosf(arg) : sinf(arg);
    }
    long o = (long)tok*Dm + d;
    unsigned short h = f2bf(v);
    x[o] = v; xbf[o] = h; membf[o] = h;
  }
}

// ---------------- bf16-WMMA GEMM: C = A * W^T + bias ----------------
// A: (M,K) bf16, W: (N,K) bf16, C: f32 or bf16. N mult of 128, K mult of 32.
// Double-buffered LDS tiles filled by pipelined global_load_async_to_lds_b128.
__global__ __launch_bounds__(256)
void nar_gemm_kernel(const unsigned short* __restrict__ A,
                     const unsigned short* __restrict__ W,
                     const float* __restrict__ bias, void* __restrict__ Cout,
                     int M, int N, int K,
                     const int* idxp, long idx_stride, int idx_add,
                     int relu, int swapLB, int out_bf16)
{
  __shared__ unsigned short lA[2][128*LDST];
  __shared__ unsigned short lB[2][128*LDST];
  if (idxp) W += ((long)(*idxp) + idx_add) * idx_stride;
  int m0 = blockIdx.x * 128;
  int n0 = blockIdx.y * 128;
  int tid = threadIdx.x;
  int lane = tid & 31, wid = tid >> 5;
  int wm = (wid & 3) * 32, wn = (wid >> 2) * 64;
  int lm = lane & 15, hh = lane >> 4;

  // per-thread async-copy plumbing: 2x16B for A, 2x16B for B per k-tile
  int lin0 = tid*8, lin1 = (tid + 256)*8;
  int r0 = lin0 >> 5, kc0 = lin0 & 31;
  int r1 = lin1 >> 5, kc1 = lin1 & 31;
  int ra0 = m0 + r0; if (ra0 >= M) ra0 = M - 1;   // clamped rows are never stored
  int ra1 = m0 + r1; if (ra1 >= M) ra1 = M - 1;
  const unsigned short* ga0 = A + (long)ra0*K + kc0;
  const unsigned short* ga1 = A + (long)ra1*K + kc1;
  const unsigned short* gb0 = W + (long)(n0 + r0)*K + kc0;
  const unsigned short* gb1 = W + (long)(n0 + r1)*K + kc1;
  unsigned dA0[2], dA1[2], dB0[2], dB1[2];
  for (int s = 0; s < 2; ++s) {
    dA0[s] = (unsigned)(size_t)&lA[s][r0*LDST + kc0];
    dA1[s] = (unsigned)(size_t)&lA[s][r1*LDST + kc1];
    dB0[s] = (unsigned)(size_t)&lB[s][r0*LDST + kc0];
    dB1[s] = (unsigned)(size_t)&lB[s][r1*LDST + kc1];
  }

  v8f c[2][4];
  for (int i = 0; i < 2; ++i) for (int j = 0; j < 4; ++j)
    for (int r = 0; r < 8; ++r) c[i][j][r] = 0.f;

  const int KT = K >> 5;
  // prologue: tile 0 -> buffer 0
  async_copy16(dA0[0], ga0);  async_copy16(dA1[0], ga1);
  async_copy16(dB0[0], gb0);  async_copy16(dB1[0], gb1);
  ga0 += 32; ga1 += 32; gb0 += 32; gb1 += 32;

  for (int kt = 0; kt < KT; ++kt) {
    int cur = kt & 1, nxt = cur ^ 1;
    if (kt + 1 < KT) {                 // prefetch next tile into other buffer
      async_copy16(dA0[nxt], ga0);  async_copy16(dA1[nxt], ga1);
      async_copy16(dB0[nxt], gb0);  async_copy16(dB1[nxt], gb1);
      ga0 += 32; ga1 += 32; gb0 += 32; gb1 += 32;
      async_wait_le4();                // in-order: current tile's 4 copies retired
    } else {
      async_wait_le0();
    }
    __syncthreads();                   // all waves' current-tile copies visible

    Frag a[2], b[4];
    const unsigned short* lAc = &lA[cur][0];
    const unsigned short* lBc = &lB[cur][0];
    // A frag: 16x32 bf16 (lanes 0-15: M=row; VGPR j -> K=h*16+2j)
    for (int i = 0; i < 2; ++i) {
      const unsigned short* p = &lAc[(wm + i*16 + lm)*LDST];
      for (int j = 0; j < 4; ++j) {
        int k = hh*16 + 2*j;
        a[i].u[2*j]     = p[k];      a[i].u[2*j+1]   = p[k+1];
        a[i].u[8+2*j]   = p[8+k];    a[i].u[8+2*j+1] = p[8+k+1];
      }
    }
    // B frag: 32x16 bf16, B[k][n] = W[n][k]; lane: N=l&15, VGPR v -> K=h*16+2v
    for (int j = 0; j < 4; ++j) {
      const unsigned short* p = &lBc[(wn + j*16 + lm)*LDST + hh*16];
      for (int v = 0; v < 8; ++v) { b[j].u[2*v] = p[2*v]; b[j].u[2*v+1] = p[2*v+1]; }
    }
    for (int i = 0; i < 2; ++i)
      for (int j = 0; j < 4; ++j)
        c[i][j] = __builtin_amdgcn_wmma_f32_16x16x32_bf16(
            false, a[i].v, false, b[j].v, (short)0, c[i][j], false, false);
    __syncthreads();                   // done reading 'cur' before it is refilled
  }
  // epilogue: C layout lane (0-15 -> M=r, 16-31 -> M=r+8), N=l&15
  for (int i = 0; i < 2; ++i) {
    int mgb = m0 + wm + i*16 + hh*8;
    for (int j = 0; j < 4; ++j) {
      int ng = n0 + wn + j*16 + lm;
      float bs = bias ? bias[ng] : 0.f;
      for (int r = 0; r < 8; ++r) {
        int m = mgb + r;
        if (m < M) {
          float v = c[i][j][r] + bs;
          if (relu) v = fmaxf(v, 0.f);
          long row;
          if (swapLB) { int ll = m / Bz, bb = m % Bz; row = (long)bb*LMAXn + ll; }
          else row = m;
          if (out_bf16) ((unsigned short*)Cout)[row*(long)N + ng] = f2bf(v);
          else          ((float*)Cout)[row*(long)N + ng] = v;
        }
      }
    }
  }
}

// ---------------- attention (bf16 in / bf16 out): one wave = 16 queries of one (b,h) ----------------
__global__ __launch_bounds__(32)
void nar_attn_kernel(const unsigned short* __restrict__ Q, int q_ld,
                     const unsigned short* __restrict__ Km, int k_ld,
                     const unsigned short* __restrict__ Vm, int v_ld,
                     unsigned short* __restrict__ O, int o_ld,
                     int Lq, int Ls)
{
  __shared__ float sc[16*SPAD];           // 51.2 KB scores, bf16 P packed in place
  int q0 = blockIdx.x * 16;
  int hh = blockIdx.y, bb = blockIdx.z;
  int lane = threadIdx.x;
  int lm = lane & 15, hl = lane >> 4;
  const float scale = 0.125f;             // 1/sqrt(64)

  Frag qa[2];
  {
    int row = q0 + lm; if (row >= Lq) row = Lq - 1;
    const unsigned short* qp = Q + ((long)row*Bz + bb)*q_ld + hh*DH;
    for (int t = 0; t < 2; ++t) {
      int kk0 = t*32;
      for (int j = 0; j < 4; ++j) {
        int k = kk0 + hl*16 + 2*j;
        qa[t].u[2*j]     = qp[k];      qa[t].u[2*j+1]   = qp[k+1];
        qa[t].u[8+2*j]   = qp[k+8];    qa[t].u[8+2*j+1] = qp[k+9];
      }
    }
  }

  // scores = Q K^T * scale, padded columns -> -1e30
  for (int st = 0; st < SPAD/16; ++st) {
    v8f cc; for (int r = 0; r < 8; ++r) cc[r] = 0.f;
    int s = st*16 + lm;
    bool sv = (s < Ls);
    int sl = sv ? s : (Ls - 1);
    const unsigned short* kp = Km + ((long)sl*Bz + bb)*k_ld + hh*DH + hl*16;
    if (st*16 < Ls) {
      Frag kb[2];
      for (int t = 0; t < 2; ++t)
        for (int v = 0; v < 8; ++v) {
          kb[t].u[2*v]   = kp[t*32 + 2*v];
          kb[t].u[2*v+1] = kp[t*32 + 2*v + 1];
        }
      cc = __builtin_amdgcn_wmma_f32_16x16x32_bf16(false, qa[0].v, false, kb[0].v, (short)0, cc, false, false);
      cc = __builtin_amdgcn_wmma_f32_16x16x32_bf16(false, qa[1].v, false, kb[1].v, (short)0, cc, false, false);
    }
    for (int r = 0; r < 8; ++r) {
      int m = r + hl*8;
      sc[m*SPAD + st*16 + lm] = sv ? cc[r]*scale : -1e30f;
    }
  }
  __syncthreads();

  // softmax per row; pack P as bf16 in place (byte 2c..2c+1 only touches f32 idx <= c)
  for (int m = 0; m < 16; ++m) {
    float* row = &sc[m*SPAD];
    float mx = -1e30f;
    for (int cI = lane; cI < SPAD; cI += 32) mx = fmaxf(mx, row[cI]);
    for (int o = 16; o > 0; o >>= 1) mx = fmaxf(mx, __shfl_xor(mx, o, 32));
    float sum = 0.f;
    for (int cI = lane; cI < SPAD; cI += 32) sum += __expf(row[cI] - mx);
    for (int o = 16; o > 0; o >>= 1) sum += __shfl_xor(sum, o, 32);
    float inv = 1.f / sum;
    unsigned short* prow = (unsigned short*)row;
    for (int cI = lane; cI < SPAD; cI += 32) {
      float p = __expf(row[cI] - mx) * inv;
      prow[cI] = f2bf(p);
    }
  }
  __syncthreads();

  // O = P V  (25 k-steps of 32, four 16-wide dh tiles)
  v8f co[4];
  for (int nt = 0; nt < 4; ++nt) for (int r = 0; r < 8; ++r) co[nt][r] = 0.f;
  for (int st = 0; st < SPAD/32; ++st) {
    Frag pa;
    const unsigned short* pr = (const unsigned short*)&sc[lm*SPAD];
    for (int j = 0; j < 4; ++j) {
      int k = st*32 + hl*16 + 2*j;
      pa.u[2*j]     = pr[k];      pa.u[2*j+1]   = pr[k+1];
      pa.u[8+2*j]   = pr[k+8];    pa.u[8+2*j+1] = pr[k+9];
    }
    for (int nt = 0; nt < 4; ++nt) {
      Frag vb;
      for (int v = 0; v < 8; ++v) {
        int s  = st*32 + hl*16 + 2*v;
        int sa = (s     < Ls) ? s     : Ls - 1;
        int sb = (s + 1 < Ls) ? s + 1 : Ls - 1;
        vb.u[2*v]   = Vm[((long)sa*Bz + bb)*v_ld + hh*DH + nt*16 + lm];
        vb.u[2*v+1] = Vm[((long)sb*Bz + bb)*v_ld + hh*DH + nt*16 + lm];
      }
      co[nt] = __builtin_amdgcn_wmma_f32_16x16x32_bf16(
          false, pa.v, false, vb.v, (short)0, co[nt], false, false);
    }
  }
  for (int nt = 0; nt < 4; ++nt)
    for (int r = 0; r < 8; ++r) {
      int row = q0 + r + hl*8;
      if (row < Lq)
        O[((long)row*Bz + bb)*o_ld + hh*DH + nt*16 + lm] = f2bf(co[nt][r]);
    }
}

// ---------------- fused residual + adaptive LayerNorm (x f32 in-place, + bf16 mirror) ----------------
__global__ __launch_bounds__(256)
void nar_adaln_kernel(float* __restrict__ x, const float* __restrict__ y,
                      const float* __restrict__ lw, const float* __restrict__ lb,
                      const int* __restrict__ idxp, unsigned short* __restrict__ xbf)
{
  __shared__ float red[2][8];
  int tok = blockIdx.x;
  int tid = threadIdx.x;
  int idx = *idxp;
  const float* w = lw + (long)idx*Dm;
  const float* b = lb + (long)idx*Dm;
  float* xp = x + (long)tok*Dm;
  const float* yp = y + (long)tok*Dm;
  float t[4], s = 0.f, s2 = 0.f;
  for (int e = 0; e < 4; ++e) {
    int d = tid + e*256;
    t[e] = xp[d] + yp[d];
    s += t[e]; s2 += t[e]*t[e];
  }
  for (int o = 16; o > 0; o >>= 1) { s += __shfl_xor(s, o, 32); s2 += __shfl_xor(s2, o, 32); }
  int wid = tid >> 5, lane = tid & 31;
  if (lane == 0) { red[0][wid] = s; red[1][wid] = s2; }
  __syncthreads();
  s = 0.f; s2 = 0.f;
  for (int i = 0; i < 8; ++i) { s += red[0][i]; s2 += red[1][i]; }
  float mu  = s / (float)Dm;
  float var = s2 / (float)Dm - mu*mu;
  float inv = rsqrtf(var + 1e-5f);
  for (int e = 0; e < 4; ++e) {
    int d = tid + e*256;
    float v = (t[e] - mu) * inv * w[d] + b[d];
    xp[d] = v;
    xbf[(long)tok*Dm + d] = f2bf(v);
  }
}

// ---------------- host orchestration ----------------
extern "C" void kernel_launch(void* const* d_in, const int* in_sizes, int n_in,
                              void* d_out, int out_size, void* d_ws, size_t ws_size,
                              hipStream_t stream) {
  (void)in_sizes; (void)n_in; (void)out_size; (void)ws_size;
  const int*   text       = (const int*)d_in[0];
  const int*   audio      = (const int*)d_in[1];
  const int*   enrolled   = (const int*)d_in[2];
  const int*   text_len   = (const int*)d_in[3];
  const int*   audio_len  = (const int*)d_in[4];
  const int*   idxp       = (const int*)d_in[5];
  const float* text_table = (const float*)d_in[6];
  const float* audio_tabs = (const float*)d_in[7];
  const float* index_tab  = (const float*)d_in[8];
  const float* sa_w_in  = (const float*)d_in[9];
  const float* sa_b_in  = (const float*)d_in[10];
  const float* sa_w_out = (const float*)d_in[11];
  const float* sa_b_out = (const float*)d_in[12];
  const float* ca_w_in  = (const float*)d_in[13];
  const float* ca_b_in  = (const float*)d_in[14];
  const float* ca_w_out = (const float*)d_in[15];
  const float* ca_b_out = (const float*)d_in[16];
  const float* ff_w1 = (const float*)d_in[17];
  const float* ff_b1 = (const float*)d_in[18];
  const float* ff_w2 = (const float*)d_in[19];
  const float* ff_b2 = (const float*)d_in[20];
  const float* ln_w  = (const float*)d_in[21];
  const float* ln_b  = (const float*)d_in[22];

  // ---- workspace carve (256B aligned chunks) ----
  char* wp = (char*)d_ws;
  auto carve = [&](size_t bytes) { char* r = wp; wp += (bytes + 255) & ~(size_t)255; return r; };
  float*          x      = (float*)          carve((size_t)MTOK*Dm*4);
  float*          bufP   = (float*)          carve((size_t)MTOK*Dm*4);
  unsigned short* xbf    = (unsigned short*) carve((size_t)MTOK*Dm*2);
  unsigned short* membf  = (unsigned short*) carve((size_t)MTOK*Dm*2);
  unsigned short* obf    = (unsigned short*) carve((size_t)MTOK*Dm*2);
  unsigned short* bigbf  = (unsigned short*) carve((size_t)MTOK*DFFm*2);  // qkv / q+kv / ff1
  unsigned short* wsaIn  = (unsigned short*) carve((size_t)NLAY*3*Dm*Dm*2);
  unsigned short* wsaOut = (unsigned short*) carve((size_t)NLAY*Dm*Dm*2);
  unsigned short* wcaIn  = (unsigned short*) carve((size_t)NLAY*3*Dm*Dm*2);
  unsigned short* wcaOut = (unsigned short*) carve((size_t)NLAY*Dm*Dm*2);
  unsigned short* wff1   = (unsigned short*) carve((size_t)NLAY*DFFm*Dm*2);
  unsigned short* wff2   = (unsigned short*) carve((size_t)NLAY*Dm*DFFm*2);
  unsigned short* wtab   = (unsigned short*) carve((size_t)NCH*AVOC*Dm*2);

  // ---- one-time (per call) weight down-conversion to bf16 ----
  auto cvt = [&](const float* src, unsigned short* dst, size_t n) {
    nar_cvt_kernel<<<(unsigned)(n/2048), 256, 0, stream>>>(src, dst);
  };
  cvt(sa_w_in,  wsaIn,  (size_t)NLAY*3*Dm*Dm);
  cvt(sa_w_out, wsaOut, (size_t)NLAY*Dm*Dm);
  cvt(ca_w_in,  wcaIn,  (size_t)NLAY*3*Dm*Dm);
  cvt(ca_w_out, wcaOut, (size_t)NLAY*Dm*Dm);
  cvt(ff_w1,    wff1,   (size_t)NLAY*DFFm*Dm);
  cvt(ff_w2,    wff2,   (size_t)NLAY*Dm*DFFm);
  cvt(audio_tabs, wtab, (size_t)NCH*AVOC*Dm);

  dim3 gAttn(SPAD/16, NHEADS, Bz);               // 50 x 16 x 4

  nar_embed_kernel<<<MTOK, 256, 0, stream>>>(text, audio, enrolled, text_len, audio_len,
                                             idxp, text_table, audio_tabs, index_tab,
                                             x, xbf, membf);
  for (int l = 0; l < NLAY; ++l) {
    // ---- self attention ----
    nar_gemm_kernel<<<dim3(25, 24), 256, 0, stream>>>(xbf, wsaIn + (size_t)l*3*Dm*Dm,
        sa_b_in + (size_t)l*3*Dm, bigbf, MTOK, 3*Dm, Dm, nullptr, 0, 0, 0, 0, 1);
    nar_attn_kernel<<<gAttn, 32, 0, stream>>>(bigbf, 3*Dm, bigbf + Dm, 3*Dm, bigbf + 2*Dm, 3*Dm,
        obf, Dm, LMAXn, LMAXn);
    nar_gemm_kernel<<<dim3(25, 8), 256, 0, stream>>>(obf, wsaOut + (size_t)l*Dm*Dm,
        sa_b_out + (size_t)l*Dm, bufP, MTOK, Dm, Dm, nullptr, 0, 0, 0, 0, 0);
    nar_adaln_kernel<<<MTOK, 256, 0, stream>>>(x, bufP,
        ln_w + ((size_t)l*3 + 0)*NCH*Dm, ln_b + ((size_t)l*3 + 0)*NCH*Dm, idxp, xbf);
    // ---- cross attention (q from x, k/v from frozen memory) ----
    unsigned short* kvbf = bigbf + (size_t)MTOK*Dm;
    nar_gemm_kernel<<<dim3(25, 8), 256, 0, stream>>>(xbf, wcaIn + (size_t)l*3*Dm*Dm,
        ca_b_in + (size_t)l*3*Dm, bigbf, MTOK, Dm, Dm, nullptr, 0, 0, 0, 0, 1);
    nar_gemm_kernel<<<dim3(25, 16), 256, 0, stream>>>(membf,
        wcaIn + (size_t)l*3*Dm*Dm + (size_t)Dm*Dm,
        ca_b_in + (size_t)l*3*Dm + Dm, kvbf, MTOK, 2*Dm, Dm, nullptr, 0, 0, 0, 0, 1);
    nar_attn_kernel<<<gAttn, 32, 0, stream>>>(bigbf, Dm, kvbf, 2*Dm, kvbf + Dm, 2*Dm,
        obf, Dm, LMAXn, LMAXn);
    nar_gemm_kernel<<<dim3(25, 8), 256, 0, stream>>>(obf, wcaOut + (size_t)l*Dm*Dm,
        ca_b_out + (size_t)l*Dm, bufP, MTOK, Dm, Dm, nullptr, 0, 0, 0, 0, 0);
    nar_adaln_kernel<<<MTOK, 256, 0, stream>>>(x, bufP,
        ln_w + ((size_t)l*3 + 1)*NCH*Dm, ln_b + ((size_t)l*3 + 1)*NCH*Dm, idxp, xbf);
    // ---- FFN ----
    nar_gemm_kernel<<<dim3(25, 32), 256, 0, stream>>>(xbf, wff1 + (size_t)l*DFFm*Dm,
        ff_b1 + (size_t)l*DFFm, bigbf, MTOK, DFFm, Dm, nullptr, 0, 0, 1, 0, 1);
    nar_gemm_kernel<<<dim3(25, 8), 256, 0, stream>>>(bigbf, wff2 + (size_t)l*Dm*DFFm,
        ff_b2 + (size_t)l*Dm, bufP, MTOK, Dm, DFFm, nullptr, 0, 0, 0, 0, 0);
    nar_adaln_kernel<<<MTOK, 256, 0, stream>>>(x, bufP,
        ln_w + ((size_t)l*3 + 2)*NCH*Dm, ln_b + ((size_t)l*3 + 2)*NCH*Dm, idxp, xbf);
  }
  // ---- final projection: out[b,l,:] = x[l,b,:] @ audio_tables[index-2].T ----
  nar_gemm_kernel<<<dim3(25, 8), 256, 0, stream>>>(xbf, wtab, nullptr, d_out,
      MTOK, AVOC, Dm, idxp, (long)AVOC*Dm, -2, 0, 1, 0);
}
